// Attention_15049565405208
// MI455X (gfx1250) — matrix operand order repaired and checked by reading, not compiled, not run on hardware
//
#include <hip/hip_runtime.h>
#include <hip/hip_bf16.h>

// ---------------------------------------------------------------------------
// CDNA5 (gfx1250) linear-attention block, bf16 WMMA pipeline with
// async global->LDS staging (ASYNCcnt) and double-buffered GEMM tiles.
// ---------------------------------------------------------------------------

typedef __attribute__((ext_vector_type(16))) __bf16 v16bf;
typedef __attribute__((ext_vector_type(8)))  __bf16 v8bf;
typedef __attribute__((ext_vector_type(8)))  float  v8f;

#define ATTN_B     8
#define ATTN_SEQ   4096
#define ATTN_DIM   768
#define ATTN_HEADS 12
#define ATTN_HD    64

__device__ __forceinline__ float elu1f(float x) {
    // elu(x)+1 == x+1 (x>=0) ; exp(x) (x<0)
    return x >= 0.0f ? x + 1.0f : __expf(x);
}

__device__ __forceinline__ v16bf cat8(v8bf lo, v8bf hi) {
    return __builtin_shufflevector(lo, hi, 0, 1, 2, 3, 4, 5, 6, 7,
                                           8, 9, 10, 11, 12, 13, 14, 15);
}

// Async DMA: 16B per lane, global -> LDS, tracked by ASYNCcnt.
__device__ __forceinline__ void async_copy_b128(void* lds_ptr, const void* gptr) {
    uint32_t lds_off = (uint32_t)(uintptr_t)lds_ptr;   // low 32 bits = LDS offset
    uint64_t gaddr   = (uint64_t)(uintptr_t)gptr;
    asm volatile("global_load_async_to_lds_b128 %0, %1, off"
                 :: "v"(lds_off), "v"(gaddr)
                 : "memory");
}

__device__ __forceinline__ void wait_async0() {
    asm volatile("s_wait_asynccnt 0" ::: "memory");
}

__device__ __forceinline__ uint32_t pack_bf16x2(float a, float b) {
    union { __bf16 h[2]; uint32_t u; } p;
    p.h[0] = (__bf16)a;
    p.h[1] = (__bf16)b;
    return p.u;
}

// ---------------------------------------------------------------------------
// Kernel 0: fp32 -> bf16 elementwise (weights)
// ---------------------------------------------------------------------------
__global__ __launch_bounds__(256)
void f32_to_bf16_kernel(const float* __restrict__ in, __bf16* __restrict__ out,
                        size_t n) {
    size_t i = (size_t)blockIdx.x * 256u + threadIdx.x;
    size_t stride = (size_t)gridDim.x * 256u;
    for (; i < n; i += stride) out[i] = (__bf16)in[i];
}

// ---------------------------------------------------------------------------
// Kernel 1: LayerNorm per row (dim=768), bf16 output
// ---------------------------------------------------------------------------
__global__ __launch_bounds__(256)
void layernorm_bf16_kernel(const float* __restrict__ x,
                           const float* __restrict__ gamma,
                           const float* __restrict__ beta,
                           __bf16* __restrict__ y, int dim) {
    const int row = blockIdx.x;
    const float* xr = x + (size_t)row * dim;
    __shared__ float s1[256];
    __shared__ float s2[256];
    const int tid = threadIdx.x;
    float a = 0.0f, b = 0.0f;
    for (int i = tid; i < dim; i += 256) {
        float v = xr[i];
        a += v;
        b += v * v;
    }
    s1[tid] = a; s2[tid] = b;
    __syncthreads();
    for (int o = 128; o > 0; o >>= 1) {
        if (tid < o) { s1[tid] += s1[tid + o]; s2[tid] += s2[tid + o]; }
        __syncthreads();
    }
    const float inv_dim = 1.0f / (float)dim;
    const float mu = s1[0] * inv_dim;
    const float var = s2[0] * inv_dim - mu * mu;
    const float rstd = rsqrtf(var + 1e-5f);
    __bf16* yr = y + (size_t)row * dim;
    for (int i = tid; i < dim; i += 256)
        yr[i] = (__bf16)(((xr[i] - mu) * rstd) * gamma[i] + beta[i]);
}

// ---------------------------------------------------------------------------
// Kernel 2/5: bf16 WMMA GEMM  C[M,N] = A[M,K] * B[N,K]^T (+bias)
// Block tile 128x128, BK=32, 256 threads = 8 waves.
// Each wave: 32x64 of C = 2x4 WMMA fragments (8 WMMA / K-step).
// Double-buffered LDS tiles filled by global_load_async_to_lds_b128.
// ---------------------------------------------------------------------------
template <bool HAS_BIAS>
__global__ __launch_bounds__(256)
void gemm_bf16_wmma_kernel(const __bf16* __restrict__ A,
                           const __bf16* __restrict__ B,
                           const float* __restrict__ bias,
                           float* __restrict__ C,
                           int M, int N, int K) {
    constexpr int BM = 128, BN = 128;
    constexpr int AS = 40;  // LDS row stride (halfs): 80B rows, 16B aligned
    constexpr int BS = 48;  // 96B rows, 32B aligned
    __shared__ __align__(32) __bf16 As[2][BM * AS];
    __shared__ __align__(32) __bf16 Bs[2][BN * BS];

    const int tid  = threadIdx.x;
    const int lane = tid & 31;
    const int wave = tid >> 5;
    const int m0 = blockIdx.x * BM;
    const int n0 = blockIdx.y * BN;
    const int wm = (wave & 3) * 32;   // wave row offset in block tile
    const int wn = (wave >> 2) * 64;  // wave col offset in block tile

    // Stage one 128x32 bf16 tile per operand: 512 chunks of 16B, 2 per thread.
    auto issue_tiles = [&](int k0, int buf) {
#pragma unroll
        for (int i = 0; i < 2; ++i) {
            const int c   = tid + 256 * i;   // 0..511
            const int r   = c >> 2;          // 0..127
            const int col = (c & 3) * 8;     // 0,8,16,24 halfs
            async_copy_b128(&As[buf][r * AS + col],
                            &A[(size_t)(m0 + r) * K + k0 + col]);
            async_copy_b128(&Bs[buf][r * BS + col],
                            &B[(size_t)(n0 + r) * K + k0 + col]);
        }
    };

    v8f acc[2][4] = {};

    issue_tiles(0, 0);
    const int nk = K >> 5;
    for (int it = 0; it < nk; ++it) {
        wait_async0();       // this wave's async writes into buf landed
        __syncthreads();     // everyone's landed; prior buf fully consumed
        if (it + 1 < nk) issue_tiles((it + 1) << 5, (it + 1) & 1);

        const __bf16* Ab = As[it & 1];
        const __bf16* Bb = Bs[it & 1];

        const int al = lane & 15;
        const int g  = (lane >> 4) * 16;  // A: contiguous K run
        const int bg = (lane >> 4) * 16;  // B: contiguous K run

        v16bf af[2];
#pragma unroll
        for (int i = 0; i < 2; ++i) {
            const int am = wm + i * 16 + al;
            v8bf lo = *reinterpret_cast<const v8bf*>(&Ab[am * AS + g]);
            v8bf hi = *reinterpret_cast<const v8bf*>(&Ab[am * AS + 8 + g]);
            af[i] = cat8(lo, hi);
        }
#pragma unroll
        for (int j = 0; j < 4; ++j) {
            v16bf bfj = *reinterpret_cast<const v16bf*>(
                &Bb[(wn + j * 16 + al) * BS + bg]);
#pragma unroll
            for (int i = 0; i < 2; ++i) {
                acc[i][j] = __builtin_amdgcn_wmma_f32_16x16x32_bf16(
                    false, af[i], false, bfj, (short)0, acc[i][j], false, false);
            }
        }
    }

    // Epilogue: VGPR r, lanes 0-15: M=r, N=lane; lanes 16-31: M=8+r, N=lane-16
    const int cn = lane & 15;
    const int cm = (lane >> 4) * 8;
#pragma unroll
    for (int i = 0; i < 2; ++i) {
#pragma unroll
        for (int j = 0; j < 4; ++j) {
#pragma unroll
            for (int r = 0; r < 8; ++r) {
                const int m = m0 + wm + i * 16 + cm + r;
                const int n = n0 + wn + j * 16 + cn;
                float o = acc[i][j][r];
                if (HAS_BIAS) o += bias[n];
                C[(size_t)m * N + n] = o;
            }
        }
    }
}

// ---------------------------------------------------------------------------
// Kernel 3: per (b,h): kv[d,e] = sum_n elu1(k[n,d]) * v[n,e]  (64x64, K=4096)
//           k_sum[d]  = sum_n elu1(k[n,d])
// Transposed staging with packed bf16x2 LDS stores. kv stored transposed
// [e][d] bf16 for the next stage's B operand.
// ---------------------------------------------------------------------------
__global__ __launch_bounds__(256)
void kv_ksum_kernel(const float* __restrict__ qkv,   // [B*SEQ, 3*DIM]
                    __bf16* __restrict__ kvT,        // [B*H, 64*64] as [e][d]
                    float* __restrict__ ksum,        // [B*H, 64]
                    int seq) {
    constexpr int AS = 40;
    constexpr int BS = 48;
    __shared__ __align__(32) __bf16 Ks[64 * AS];  // [d][n] transposed
    __shared__ __align__(32) __bf16 Vs[64 * BS];  // [e][n] transposed
    __shared__ float ks_lds[64];

    const int bh = blockIdx.x;
    const int b = bh / ATTN_HEADS, h = bh % ATTN_HEADS;
    const float* kbase = qkv + ((size_t)b * seq) * (3 * ATTN_DIM) + ATTN_DIM + h * ATTN_HD;
    const float* vbase = qkv + ((size_t)b * seq) * (3 * ATTN_DIM) + 2 * ATTN_DIM + h * ATTN_HD;

    const int tid  = threadIdx.x;
    const int lane = tid & 31;
    const int wave = tid >> 5;
    if (tid < 64) ks_lds[tid] = 0.0f;

    // staging role: token pair (2p, 2p+1), 4 head-dims starting at d0
    const int p  = tid & 15;         // token pair id within 32-chunk
    const int d0 = (tid >> 4) * 4;   // 0..60

    float kacc[4];
#pragma unroll
    for (int j = 0; j < 4; ++j) kacc[j] = 0.0f;

    const int wm = (wave & 3) * 16;   // d offset of wave's frags
    const int wn = (wave >> 2) * 32;  // e offset
    v8f acc0 = {};
    v8f acc1 = {};

    for (int t0 = 0; t0 < seq; t0 += 32) {
        __syncthreads();
        const float* kr0 = kbase + (size_t)(t0 + 2 * p) * (3 * ATTN_DIM) + d0;
        const float* kr1 = kr0 + 3 * ATTN_DIM;
        const float* vr0 = vbase + (size_t)(t0 + 2 * p) * (3 * ATTN_DIM) + d0;
        const float* vr1 = vr0 + 3 * ATTN_DIM;
#pragma unroll
        for (int j = 0; j < 4; ++j) {
            const float k0 = elu1f(kr0[j]);
            const float k1 = elu1f(kr1[j]);
            kacc[j] += k0 + k1;
            *reinterpret_cast<uint32_t*>(&Ks[(d0 + j) * AS + 2 * p]) =
                pack_bf16x2(k0, k1);
            *reinterpret_cast<uint32_t*>(&Vs[(d0 + j) * BS + 2 * p]) =
                pack_bf16x2(vr0[j], vr1[j]);
        }
        __syncthreads();

        const int am = wm + (lane & 15);
        const int g  = (lane >> 4) * 16;
        v8bf alo = *reinterpret_cast<const v8bf*>(&Ks[am * AS + g]);
        v8bf ahi = *reinterpret_cast<const v8bf*>(&Ks[am * AS + 8 + g]);
        v16bf af = cat8(alo, ahi);

        const int bn = lane & 15;
        const int bg = (lane >> 4) * 16;
        v16bf b0 = *reinterpret_cast<const v16bf*>(&Vs[(bn + wn) * BS + bg]);
        v16bf b1 = *reinterpret_cast<const v16bf*>(&Vs[(bn + wn + 16) * BS + bg]);

        acc0 = __builtin_amdgcn_wmma_f32_16x16x32_bf16(false, af, false, b0,
                                                       (short)0, acc0, false, false);
        acc1 = __builtin_amdgcn_wmma_f32_16x16x32_bf16(false, af, false, b1,
                                                       (short)0, acc1, false, false);
    }

#pragma unroll
    for (int j = 0; j < 4; ++j) atomicAdd(&ks_lds[d0 + j], kacc[j]);
    __syncthreads();
    if (tid < 64) ksum[(size_t)bh * 64 + tid] = ks_lds[tid];

    // write kv transposed [e][d] bf16
    const int cn = lane & 15;
    const int cm = (lane >> 4) * 8;
    __bf16* dst = kvT + (size_t)bh * (64 * 64);
#pragma unroll
    for (int r = 0; r < 8; ++r) {
        const int d  = wm + cm + r;
        const int e1 = wn + cn;
        const int e2 = e1 + 16;
        dst[e1 * 64 + d] = (__bf16)acc0[r];
        dst[e2 * 64 + d] = (__bf16)acc1[r];
    }
}

// ---------------------------------------------------------------------------
// Kernel 4: per (b,h, 128-token tile): out[n,e] = z[n] * sum_d elu1(q[n,d])*kv[d,e]
//           z[n] = 1/(elu1(q[n,:]) . ksum + eps)
// kv tile pulled with async global->LDS DMA; q staged as packed v8bf.
// Output bf16 in [b, n, h*64+e] layout (A operand of projection GEMM).
// ---------------------------------------------------------------------------
__global__ __launch_bounds__(256)
void attn_out_kernel(const float* __restrict__ qkv,   // [B*SEQ, 3*DIM]
                     const __bf16* __restrict__ kvT,  // [B*H][e][d]
                     const float* __restrict__ ksum,  // [B*H][64]
                     __bf16* __restrict__ attn,       // [B*SEQ, DIM]
                     int seq) {
    constexpr int AS = 80;  // 160B rows -> 32B aligned
    constexpr int BS = 80;
    __shared__ __align__(32) __bf16 Qs[128 * AS];  // [n_local][d]
    __shared__ __align__(32) __bf16 KVs[64 * BS];  // [e][d]
    __shared__ float ks_lds[64];
    __shared__ float z_lds[128];

    const int bh = blockIdx.y;
    const int b = bh / ATTN_HEADS, h = bh % ATTN_HEADS;
    const int n0 = blockIdx.x * 128;
    const int tid  = threadIdx.x;
    const int lane = tid & 31;
    const int wave = tid >> 5;

    // async DMA kv tile into LDS (overlaps the q staging below)
    {
        const __bf16* src = kvT + (size_t)bh * (64 * 64);
#pragma unroll
        for (int i = 0; i < 2; ++i) {
            const int c   = tid + 256 * i;   // 0..511 chunks of 8 halfs
            const int e   = c >> 3;          // 0..63
            const int col = (c & 7) * 8;     // 0..56
            async_copy_b128(&KVs[e * BS + col], &src[e * 64 + col]);
        }
    }

    const float* qbase = qkv + ((size_t)(b * seq + n0)) * (3 * ATTN_DIM) + h * ATTN_HD;
    {   // stage elu1(q) tile: 128 x 64, packed 16B LDS stores
        const int row = tid >> 1;
        const int dc  = (tid & 1) * 32;
        const float* qr = qbase + (size_t)row * (3 * ATTN_DIM) + dc;
#pragma unroll
        for (int g8 = 0; g8 < 32; g8 += 8) {
            v8bf t;
#pragma unroll
            for (int j = 0; j < 8; ++j) t[j] = (__bf16)elu1f(qr[g8 + j]);
            *reinterpret_cast<v8bf*>(&Qs[row * AS + dc + g8]) = t;
        }
    }
    if (tid < 64) ks_lds[tid] = ksum[(size_t)bh * 64 + tid];
    wait_async0();
    __syncthreads();

    // z per local row
    if (tid < 128) {
        float dot = 0.0f;
#pragma unroll
        for (int j = 0; j < 64; ++j) dot += (float)Qs[tid * AS + j] * ks_lds[j];
        z_lds[tid] = 1.0f / (dot + 1e-6f);
    }

    // each wave: rows [wave*16, wave*16+16), all 64 cols -> 4 frags, K=64
    v8f acc[4] = {};
    const int wm = wave * 16;
#pragma unroll
    for (int kk = 0; kk < 64; kk += 32) {
        const int am = wm + (lane & 15);
        const int g  = (lane >> 4) * 16;
        v8bf alo = *reinterpret_cast<const v8bf*>(&Qs[am * AS + kk + g]);
        v8bf ahi = *reinterpret_cast<const v8bf*>(&Qs[am * AS + kk + 8 + g]);
        v16bf af = cat8(alo, ahi);
        const int bn = lane & 15;
        const int bg = (lane >> 4) * 16;
#pragma unroll
        for (int nf = 0; nf < 4; ++nf) {
            v16bf bfj = *reinterpret_cast<const v16bf*>(
                &KVs[(nf * 16 + bn) * BS + kk + bg]);
            acc[nf] = __builtin_amdgcn_wmma_f32_16x16x32_bf16(
                false, af, false, bfj, (short)0, acc[nf], false, false);
        }
    }
    __syncthreads();  // z_lds visible

    const int cn = lane & 15;
    const int cm = (lane >> 4) * 8;
#pragma unroll
    for (int r = 0; r < 8; ++r) {
        const int ml = wm + cm + r;  // 0..127
        const float z = z_lds[ml];
        const int n = n0 + ml;
        __bf16* orow = attn + ((size_t)(b * seq + n)) * ATTN_DIM + h * ATTN_HD;
#pragma unroll
        for (int nf = 0; nf < 4; ++nf) {
            const int e = nf * 16 + cn;
            orow[e] = (__bf16)(acc[nf][r] * z);
        }
    }
}

// ---------------------------------------------------------------------------
// Host-side launcher
// ---------------------------------------------------------------------------
extern "C" void kernel_launch(void* const* d_in, const int* in_sizes, int n_in,
                              void* d_out, int out_size, void* d_ws, size_t ws_size,
                              hipStream_t stream) {
    (void)in_sizes; (void)n_in; (void)out_size; (void)ws_size;

    const float* x     = (const float*)d_in[0];
    const float* gamma = (const float*)d_in[1];
    const float* beta  = (const float*)d_in[2];
    const float* wqkv  = (const float*)d_in[3];  // [3*DIM, DIM] = [N,K]
    const float* wproj = (const float*)d_in[4];  // [DIM, DIM]   = [N,K]
    const float* bproj = (const float*)d_in[5];
    float* out = (float*)d_out;

    const int B = ATTN_B, SEQ = ATTN_SEQ, DIM = ATTN_DIM, H = ATTN_HEADS;
    const int M = B * SEQ;  // 32768

    auto align_up = [](size_t v) { return (v + 255) & ~(size_t)255; };
    char* ws = (char*)d_ws;
    __bf16* y_bf    = (__bf16*)ws; ws += align_up((size_t)M * DIM * sizeof(__bf16));
    __bf16* wqkv_bf = (__bf16*)ws; ws += align_up((size_t)3 * DIM * DIM * sizeof(__bf16));
    __bf16* wproj_bf= (__bf16*)ws; ws += align_up((size_t)DIM * DIM * sizeof(__bf16));
    float*  qkv     = (float*)ws;  ws += align_up((size_t)M * 3 * DIM * sizeof(float));
    __bf16* kvT     = (__bf16*)ws; ws += align_up((size_t)B * H * 64 * 64 * sizeof(__bf16));
    float*  ksum    = (float*)ws;  ws += align_up((size_t)B * H * 64 * sizeof(float));
    __bf16* attn    = (__bf16*)ws; ws += align_up((size_t)M * DIM * sizeof(__bf16));

    // 0) weight conversion
    f32_to_bf16_kernel<<<512, 256, 0, stream>>>(wqkv, wqkv_bf, (size_t)3 * DIM * DIM);
    f32_to_bf16_kernel<<<256, 256, 0, stream>>>(wproj, wproj_bf, (size_t)DIM * DIM);

    // 1) LayerNorm -> bf16
    layernorm_bf16_kernel<<<M, 256, 0, stream>>>(x, gamma, beta, y_bf, DIM);

    // 2) QKV GEMM: [32768 x 2304] = y_bf * wqkv^T
    {
        dim3 grid(M / 128, (3 * DIM) / 128);   // 256 x 18
        gemm_bf16_wmma_kernel<false><<<grid, 256, 0, stream>>>(
            y_bf, wqkv_bf, nullptr, qkv, M, 3 * DIM, DIM);
    }

    // 3) kv + k_sum per (b,h)
    kv_ksum_kernel<<<B * H, 256, 0, stream>>>(qkv, kvT, ksum, SEQ);

    // 4) q·kv scaled by z, per (b,h, 128 tokens)
    {
        dim3 grid(SEQ / 128, B * H);           // 32 x 96
        attn_out_kernel<<<grid, 256, 0, stream>>>(qkv, kvT, ksum, attn, SEQ);
    }

    // 5) projection GEMM + bias -> fp32 out
    {
        dim3 grid(M / 128, DIM / 128);         // 256 x 6
        gemm_bf16_wmma_kernel<true><<<grid, 256, 0, stream>>>(
            attn, wproj_bf, bproj, out, M, DIM, DIM);
    }
}
